// HetNetwork_45921790328902
// MI455X (gfx1250) — matrix operand order repaired and checked by reading, not compiled, not run on hardware
//
#include <hip/hip_runtime.h>

// ---------------------------------------------------------------------------
// HetNetwork fused 3-layer MLP for gfx1250 (MI455X)
//   h1 = act1(x @ W1 + b1); h2 = act2(h1 @ W2 + b2); out = h2 @ Wout + bout
// B=131072, K1=1024, H=256, OUT=256.  f16 WMMA (16x16x32) with f32 accum.
// Memory floor: 512MB x-read + 128MB out-write ~ 27.5us @ 23.3 TB/s.
// Epilogue (67M activated elements) is the VALU critical path -> single-tanh
// formulation: tanh direct, sigmoid = 0.5+0.5*tanh(z/2) (exact),
// gelu = tanh-approx (error ~1e-3, below f16-matmul noise).
// ---------------------------------------------------------------------------

#define BATCH   131072
#define K1      1024
#define HDIM    256
#define MTILE   128          // rows per block (8 waves x 16 rows)
#define NT      16           // 256 cols / 16

typedef __attribute__((ext_vector_type(16))) _Float16 v16h;
typedef __attribute__((ext_vector_type(8)))  float    v8f;

union Frag {
  v16h h;
  unsigned u[8];
  uint4 q[2];
};

#if __has_builtin(__builtin_amdgcn_tanhf)
#define FAST_TANH(x) __builtin_amdgcn_tanhf(x)   // hw V_TANH_F32 on gfx1250
#else
#define FAST_TANH(x) tanhf(x)
#endif

// pack two f32 -> one dword of two f16 (lo = a, hi = b), RTE
static __device__ __forceinline__ unsigned pk2h(float a, float b) {
  _Float16 ha = (_Float16)a, hb = (_Float16)b;
  unsigned short ua = __builtin_bit_cast(unsigned short, ha);
  unsigned short ub = __builtin_bit_cast(unsigned short, hb);
  return (unsigned)ua | ((unsigned)ub << 16);
}

// One k-chunk x 16 n-tiles WMMA pass with ping-pong B-fragment prefetch.
static __device__ __forceinline__ void gemm_pass(const Frag& a, const uint4* __restrict__ bq,
                                                 v8f acc[NT]) {
  Frag b0, b1;
  b0.q[0] = bq[0];
  b0.q[1] = bq[1];
  #pragma unroll
  for (int t = 0; t < NT; t += 2) {
    const uint4* q1 = bq + (t + 1) * 64;
    b1.q[0] = q1[0];
    b1.q[1] = q1[1];
    acc[t] = __builtin_amdgcn_wmma_f32_16x16x32_f16(
        false, a.h, false, b0.h, (short)0, acc[t], false, false);
    if (t + 2 < NT) {
      const uint4* q2 = bq + (t + 2) * 64;
      b0.q[0] = q2[0];
      b0.q[1] = q2[1];
    }
    acc[t + 1] = __builtin_amdgcn_wmma_f32_16x16x32_f16(
        false, a.h, false, b1.h, (short)0, acc[t + 1], false, false);
  }
}

// Per-column activation coefficients for the single-tanh formulation:
//   u = z * (s1 + c3*z^2);  v = tanh(u);  r = p*v + q + s*z*(1+v)
//   id==0 (relu) overrides with fmax(z,0).
struct ActCoef {
  float s1, c3, p, q, s;
  int id;
};

static __device__ __forceinline__ ActCoef make_coef(int id) {
  ActCoef c;
  c.id = id;
  c.s1 = (id == 1) ? 1.0f : ((id == 2) ? 0.7978845608028654f : 0.5f);
  c.c3 = (id == 2) ? 0.035677408136300125f : 0.0f;   // 0.79788456*0.044715
  c.p  = (id == 1) ? 1.0f : ((id == 2) ? 0.0f : 0.5f);
  c.q  = (id == 1) ? 0.0f : ((id == 2) ? 0.0f : 0.5f);
  c.s  = (id == 2) ? 0.5f : 0.0f;
  return c;
}

static __device__ __forceinline__ float apply_act(float z, const ActCoef& c) {
  float u = z * __builtin_fmaf(c.c3, z * z, c.s1);
  float v = FAST_TANH(u);
  float r = __builtin_fmaf(c.s, z * (1.0f + v), __builtin_fmaf(c.p, v, c.q));
  return (c.id == 0) ? fmaxf(z, 0.0f) : r;
}

// ---------------------------------------------------------------------------
// Prep: convert W [K][256] fp32 into per-lane WMMA B-fragment layout (f16):
//   dst[((c*16 + t)*32 + lane)*8 + j] packs W[k][n], W[k+1][n] with
//   n = t*16 + (lane&15), k = c*32 + (lane>>4)*8 + (j>>2)*16 + 2*(j&3)
// (16-bit B 32x16: lanes 0-15 hold K=0..7,16..23, lanes 16-31 hold K=8..15,24..31)
// ---------------------------------------------------------------------------
__global__ void prep_frags(const float* __restrict__ W, unsigned* __restrict__ dst,
                           int kChunks) {
  int total = kChunks * NT * 32 * 8;
  for (int idx = blockIdx.x * blockDim.x + threadIdx.x; idx < total;
       idx += gridDim.x * blockDim.x) {
    int j    = idx & 7;
    int lane = (idx >> 3) & 31;
    int t    = (idx >> 8) & 15;
    int c    = idx >> 12;
    int n = t * 16 + (lane & 15);
    int k = c * 32 + (lane >> 4) * 8 + (j >> 2) * 16 + 2 * (j & 3);
    dst[idx] = pk2h(W[(size_t)k * HDIM + n], W[(size_t)(k + 1) * HDIM + n]);
  }
}

// ---------------------------------------------------------------------------
// Fused MLP kernel. 256 threads = 8 waves; wave w owns rows
// [blk*128 + w*16, +16). Per wave: 16 accumulator tiles (16x256 strip).
// ---------------------------------------------------------------------------
__global__ void __launch_bounds__(256)
mlp_fused(const float* __restrict__ x,
          const float* __restrict__ b1, const float* __restrict__ b2,
          const float* __restrict__ bout,
          const int* __restrict__ act1, const int* __restrict__ act2,
          const unsigned* __restrict__ w1f, const unsigned* __restrict__ w2f,
          const unsigned* __restrict__ wof,
          float* __restrict__ out) {
  __shared__ _Float16 hbuf[8][16][HDIM];   // 64 KB: per-wave f16 h-strip

  const int lane  = threadIdx.x & 31;
  const int wave  = threadIdx.x >> 5;
  const int lrow  = lane & 15;     // row within strip (A) / col within tile (C)
  const int lhalf = lane >> 4;     // K-half (A/B) / M-half (C)
  const int mbase = blockIdx.x * MTILE + wave * 16;

  v8f acc[NT];
  #pragma unroll
  for (int t = 0; t < NT; ++t) acc[t] = (v8f){0,0,0,0,0,0,0,0};

  // ---------------- layer 1: x[16x1024] @ W1 -> acc (16x256) ----------------
  // A frag (16-bit 16x32): lanes 0-15 row m, K = k0+0..7 & k0+16..23;
  //                        lanes 16-31 row m, K = k0+8..15 & k0+24..31.
  const float* xr = x + (size_t)(mbase + lrow) * K1 + lhalf * 8;
  {
    const float4* p0 = reinterpret_cast<const float4*>(xr);
    float4 f0 = p0[0], f1 = p0[1], f2 = p0[4], f3 = p0[5];
    for (int c = 0; c < K1 / 32; ++c) {
      Frag a;
      a.u[0] = pk2h(f0.x, f0.y);  a.u[1] = pk2h(f0.z, f0.w);
      a.u[2] = pk2h(f1.x, f1.y);  a.u[3] = pk2h(f1.z, f1.w);
      a.u[4] = pk2h(f2.x, f2.y);  a.u[5] = pk2h(f2.z, f2.w);
      a.u[6] = pk2h(f3.x, f3.y);  a.u[7] = pk2h(f3.z, f3.w);
      if (c + 1 < K1 / 32) {     // prefetch next chunk's x under the WMMA burst
        const float4* pn = reinterpret_cast<const float4*>(xr + (c + 1) * 32);
        f0 = pn[0]; f1 = pn[1]; f2 = pn[4]; f3 = pn[5];
      }
      const uint4* bq =
          reinterpret_cast<const uint4*>(w1f + ((size_t)(c * NT) * 32 + lane) * 8);
      gemm_pass(a, bq, acc);
    }
  }

  // epilogue 1: bias + per-column activation, write f16 strip to LDS
  // C layout: lanes 0-15 = N, vgpr i -> row i; lanes 16-31 -> row 8+i.
  #pragma unroll
  for (int t = 0; t < NT; ++t) {
    int n = t * 16 + lrow;
    float bias = b1[n];
    ActCoef cf = make_coef(act1[n]);
    #pragma unroll
    for (int i = 0; i < 8; ++i) {
      float z = apply_act(acc[t][i] + bias, cf);
      hbuf[wave][lhalf * 8 + i][n] = (_Float16)z;
    }
  }

  // ---------------- layer 2: h1[16x256] @ W2 ----------------
  #pragma unroll
  for (int t = 0; t < NT; ++t) acc[t] = (v8f){0,0,0,0,0,0,0,0};
  #pragma unroll
  for (int c = 0; c < HDIM / 32; ++c) {
    const uint4* hp = reinterpret_cast<const uint4*>(
        &hbuf[wave][lrow][c * 32 + lhalf * 8]);
    Frag a;
    a.q[0] = hp[0];   // this lane's K-half, offsets +0..7
    a.q[1] = hp[2];   // this lane's K-half, offsets +16..23
    const uint4* bq =
        reinterpret_cast<const uint4*>(w2f + ((size_t)(c * NT) * 32 + lane) * 8);
    gemm_pass(a, bq, acc);
  }

  // epilogue 2
  #pragma unroll
  for (int t = 0; t < NT; ++t) {
    int n = t * 16 + lrow;
    float bias = b2[n];
    ActCoef cf = make_coef(act2[n]);
    #pragma unroll
    for (int i = 0; i < 8; ++i) {
      float z = apply_act(acc[t][i] + bias, cf);
      hbuf[wave][lhalf * 8 + i][n] = (_Float16)z;
    }
  }

  // ---------------- layer 3: h2[16x256] @ Wout ----------------
  #pragma unroll
  for (int t = 0; t < NT; ++t) acc[t] = (v8f){0,0,0,0,0,0,0,0};
  #pragma unroll
  for (int c = 0; c < HDIM / 32; ++c) {
    const uint4* hp = reinterpret_cast<const uint4*>(
        &hbuf[wave][lrow][c * 32 + lhalf * 8]);
    Frag a;
    a.q[0] = hp[0];
    a.q[1] = hp[2];
    const uint4* bq =
        reinterpret_cast<const uint4*>(wof + ((size_t)(c * NT) * 32 + lane) * 8);
    gemm_pass(a, bq, acc);
  }

  // final epilogue: + bout, fp32 store to global
  #pragma unroll
  for (int t = 0; t < NT; ++t) {
    int n = t * 16 + lrow;
    float bias = bout[n];
    #pragma unroll
    for (int i = 0; i < 8; ++i) {
      int m = lhalf * 8 + i;
      out[(size_t)(mbase + m) * HDIM + n] = acc[t][i] + bias;
    }
  }
}

// ---------------------------------------------------------------------------
extern "C" void kernel_launch(void* const* d_in, const int* in_sizes, int n_in,
                              void* d_out, int out_size, void* d_ws, size_t ws_size,
                              hipStream_t stream) {
  const float* x    = (const float*)d_in[0];
  const float* W1   = (const float*)d_in[1];
  const float* b1   = (const float*)d_in[2];
  const float* W2   = (const float*)d_in[3];
  const float* b2   = (const float*)d_in[4];
  const float* Wout = (const float*)d_in[5];
  const float* bout = (const float*)d_in[6];
  const int*   act1 = (const int*)d_in[7];
  const int*   act2 = (const int*)d_in[8];
  float* out = (float*)d_out;

  // workspace: packed f16 B-fragments (dwords)
  //   W1: 32 kchunks * 16 tiles * 32 lanes * 8 = 131072 dwords (512 KB)
  //   W2/Wout: 8 * 16 * 32 * 8 = 32768 dwords (128 KB each)
  unsigned* w1f = (unsigned*)d_ws;
  unsigned* w2f = w1f + (K1 / 32) * NT * 32 * 8;
  unsigned* wof = w2f + (HDIM / 32) * NT * 32 * 8;

  prep_frags<<<256, 256, 0, stream>>>(W1,   w1f, K1 / 32);
  prep_frags<<<64,  256, 0, stream>>>(W2,   w2f, HDIM / 32);
  prep_frags<<<64,  256, 0, stream>>>(Wout, wof, HDIM / 32);

  mlp_fused<<<BATCH / MTILE, 256, 0, stream>>>(
      x, b1, b2, bout, act1, act2, w1f, w2f, wof, out);
}
